// GRUModel_6107443495613
// MI455X (gfx1250) — matrix operand order repaired and checked by reading, not compiled, not run on hardware
//
#include <hip/hip_runtime.h>
#include <hip/hip_bf16.h>

// ---------------- problem constants ----------------
#define BATCH 64
#define TSEQ  512
#define DIN   968
#define DPAD  992          // 31 * 32, zero-padded K for bf16 WMMA
#define HID   128
#define G3    384          // 3 * HID
#define BT    (BATCH*TSEQ) // 32768

typedef __bf16 bf16;
typedef __attribute__((ext_vector_type(8)))  __bf16 bf16x8;
typedef __attribute__((ext_vector_type(16))) __bf16 bf16x16;
typedef __attribute__((ext_vector_type(8)))  float  f32x8;

__device__ __forceinline__ bf16x16 make_frag(bf16x8 lo, bf16x8 hi) {
  return __builtin_shufflevector(lo, hi, 0,1,2,3,4,5,6,7,8,9,10,11,12,13,14,15);
}

// A fragment (16x32 bf16, row-major source, row stride ld elements).
// CDNA5 16-bit A layout: lanes 0-15 hold K{0..7,16..23}, lanes 16-31 K{8..15,24..31}.
__device__ __forceinline__ bf16x16 load_a_frag(const bf16* p, int ld, int lane) {
  int half = lane >> 4, r = lane & 15;
  const bf16* row = p + (size_t)r * (size_t)ld;
  bf16x8 lo = *(const bf16x8*)(row + 8*half);       // K 0..7  | 8..15
  bf16x8 hi = *(const bf16x8*)(row + 16 + 8*half);  // K 16..23| 24..31
  return make_frag(lo, hi);
}

// B fragment (32x16 bf16). B[k][n] = W[n][k] with W row-major [N][K]:
// lane&15 = n, elements are contiguous K; lanes 16-31 take K 16..31.
__device__ __forceinline__ bf16x16 load_b_frag(const bf16* p, int ld, int lane) {
  int half = lane >> 4, r = lane & 15;
  const bf16* row = p + (size_t)r * (size_t)ld + 16*half;
  bf16x8 lo = *(const bf16x8*)(row);
  bf16x8 hi = *(const bf16x8*)(row + 8);
  return make_frag(lo, hi);
}

__device__ __forceinline__ f32x8 wmma_bf16(bf16x16 a, bf16x16 b, f32x8 c) {
  return __builtin_amdgcn_wmma_f32_16x16x32_bf16(false, a, false, b, (short)0, c,
                                                 false, false);
}

__device__ __forceinline__ float sigmoidf_(float x) {
  return 1.0f / (1.0f + __expf(-x));
}

// ---------------- kernel 0: fp32 -> bf16 (with K zero-pad) ----------------
__global__ void conv_bf16_pad(const float* __restrict__ src, bf16* __restrict__ dst,
                              int rows, int srcK, int dstK) {
  size_t i = (size_t)blockIdx.x * blockDim.x + threadIdx.x;
  size_t total = (size_t)rows * (size_t)dstK;
  if (i >= total) return;
  int r = (int)(i / dstK);
  int k = (int)(i - (size_t)r * dstK);
  dst[i] = (k < srcK) ? (bf16)src[(size_t)r * srcK + k] : (bf16)0.0f;
}

// ---------------- kernel 1: xp = x @ w_ih^T + b_ih (both dirs) -------------
// grid: (BT/16, 3, 2); block: 256 (8 wave32s, one 16x16 N-tile per wave)
__global__ void xp_gemm(const bf16* __restrict__ xb,
                        const bf16* __restrict__ wihf, const bf16* __restrict__ wihb,
                        const float* __restrict__ bihf, const float* __restrict__ bihb,
                        float* __restrict__ xp) {
  const int mtile = blockIdx.x;                  // 0..2047
  const int wave  = threadIdx.x >> 5;            // 0..7
  const int lane  = threadIdx.x & 31;
  const int ntile = blockIdx.y * 8 + wave;       // 0..23
  const int dir   = blockIdx.z;

  const bf16*  W    = dir ? wihb : wihf;
  const float* bias = dir ? bihb : bihf;

  const int half = lane >> 4, r = lane & 15;
  const bf16* pa = xb + (size_t)(mtile * 16 + r) * DPAD;          // x row (M)
  const bf16* pb = W  + (size_t)(ntile * 16 + r) * DPAD + 16*half; // W row (N)

  f32x8 acc = {};
  for (int k0 = 0; k0 < DPAD; k0 += 32) {
    if (k0 + 128 < DPAD) {
      __builtin_prefetch(pa + k0 + 128, 0, 1);
      __builtin_prefetch(pb + k0 + 128, 0, 1);
    }
    bf16x8 alo = *(const bf16x8*)(pa + k0 + 16*half);
    bf16x8 ahi = *(const bf16x8*)(pa + k0 + 16*half + 8);
    bf16x8 blo = *(const bf16x8*)(pb + k0);
    bf16x8 bhi = *(const bf16x8*)(pb + k0 + 8);
    acc = wmma_bf16(make_frag(alo, ahi), make_frag(blo, bhi), acc);
  }

  const int   col = ntile * 16 + r;
  const float bv  = bias[col];
  float* outp = xp + (size_t)dir * ((size_t)BT * G3);
  #pragma unroll
  for (int v = 0; v < 8; ++v) {
    int m = mtile * 16 + v + 8*half;
    outp[(size_t)m * G3 + col] = acc[v] + bv;
  }
}

// ---------------- kernel 2: persistent GRU scan ---------------------------
// grid: 8 WGs (2 dirs x 4 batch-blocks of 16), block 256 (8 wave32s)
#define SCAN_SMEM (G3*HID*2 + 16*HID*2 + 16*HID*4 + 16*G3*4 + G3*4) // 136704 B

__global__ void gru_scan(const bf16* __restrict__ whhf, const bf16* __restrict__ whhb,
                         const float* __restrict__ bhhf, const float* __restrict__ bhhb,
                         const float* __restrict__ xp, float* __restrict__ out) {
  extern __shared__ char smem[];
  bf16*  wsh = (bf16*)smem;                                  // [384][128] bf16
  bf16*  hbf = (bf16*)(smem + G3*HID*2);                     // [16][128] bf16
  float* hf  = (float*)(smem + G3*HID*2 + 16*HID*2);         // [16][128] f32
  float* gh  = (float*)(smem + G3*HID*2 + 16*HID*2 + 16*HID*4);      // [16][384]
  float* bsh = (float*)(smem + G3*HID*2 + 16*HID*2 + 16*HID*4 + 16*G3*4); // [384]

  const int dir  = blockIdx.x >> 2;
  const int b0   = (blockIdx.x & 3) * 16;
  const int tid  = threadIdx.x;
  const int wave = tid >> 5, lane = tid & 31;
  const int half = lane >> 4, r = lane & 15;

  const bf16*  Wg = dir ? whhb : whhf;
  const float* bg = dir ? bhhb : bhhf;

  // stage w_hh (bf16) + b_hh into LDS, zero h
  for (int i = tid; i < (G3*HID*2)/16; i += 256)
    ((int4*)wsh)[i] = ((const int4*)Wg)[i];
  for (int i = tid; i < G3; i += 256) bsh[i] = bg[i];
  for (int i = tid; i < 16*HID; i += 256) { hf[i] = 0.0f; hbf[i] = (bf16)0.0f; }
  __syncthreads();

  const int eb = tid >> 4;            // local batch row for elementwise phase
  const int j0 = (tid & 15) * 8;      // 8 hidden channels per thread
  const float* xpd = xp + (size_t)dir * ((size_t)BT * G3);

  for (int t = 0; t < TSEQ; ++t) {
    const int tt = dir ? (TSEQ - 1 - t) : t;

    // early gi loads for this step (overlap with WMMA phase)
    const float* gi = xpd + ((size_t)(b0 + eb) * TSEQ + tt) * G3 + j0;
    float gir[8], giz[8], gin[8];
    #pragma unroll
    for (int u = 0; u < 8; ++u) {
      gir[u] = gi[u];
      giz[u] = gi[128 + u];
      gin[u] = gi[256 + u];
    }

    // gh = h @ w_hh^T + b_hh : 3 N-tiles per wave, K = 128 = 4 x 32
    const bf16* pa = hbf + (size_t)r * HID;
    #pragma unroll
    for (int jt = 0; jt < 3; ++jt) {
      const int nt = wave * 3 + jt;
      const bf16* pb = wsh + (size_t)(nt * 16 + r) * HID + 16*half;
      f32x8 acc = {};
      #pragma unroll
      for (int k0 = 0; k0 < HID; k0 += 32) {
        bf16x8 alo = *(const bf16x8*)(pa + k0 + 16*half);
        bf16x8 ahi = *(const bf16x8*)(pa + k0 + 16*half + 8);
        bf16x8 blo = *(const bf16x8*)(pb + k0);
        bf16x8 bhi = *(const bf16x8*)(pb + k0 + 8);
        acc = wmma_bf16(make_frag(alo, ahi), make_frag(blo, bhi), acc);
      }
      const int col = nt * 16 + r;
      const float bv = bsh[col];
      #pragma unroll
      for (int v = 0; v < 8; ++v)
        gh[(size_t)(v + 8*half) * G3 + col] = acc[v] + bv;
    }
    __syncthreads();

    // fused gate update
    const float* ghrow = gh + (size_t)eb * G3;
    float* hrow = hf + (size_t)eb * HID;
    bf16*  brow = hbf + (size_t)eb * HID;
    float* orow = out + ((size_t)(b0 + eb) * TSEQ + tt) * (2*HID) + dir * HID + j0;
    #pragma unroll
    for (int u = 0; u < 8; ++u) {
      const int j = j0 + u;
      const float rr = sigmoidf_(gir[u] + ghrow[j]);
      const float zz = sigmoidf_(giz[u] + ghrow[128 + j]);
      const float nn = tanhf(gin[u] + rr * ghrow[256 + j]);
      const float hn = (1.0f - zz) * nn + zz * hrow[j];
      hrow[j] = hn;
      brow[j] = (bf16)hn;
      orow[u] = hn;
    }
    __syncthreads();
  }
}

// ---------------- kernel 3: LayerNorm (unbiased std, clamped), in-place ----
__global__ void layernorm_rows(float* __restrict__ out,
                               const float* __restrict__ lnw,
                               const float* __restrict__ lnb) {
  const int row  = blockIdx.x * 8 + (threadIdx.x >> 5);  // one wave32 per row
  const int lane = threadIdx.x & 31;
  float* p = out + (size_t)row * (2*HID);

  float v[8], s = 0.0f, s2 = 0.0f;
  #pragma unroll
  for (int u = 0; u < 8; ++u) {
    v[u] = p[lane + u*32];
    s  += v[u];
    s2 += v[u] * v[u];
  }
  #pragma unroll
  for (int off = 16; off > 0; off >>= 1) {
    s  += __shfl_xor(s,  off, 32);
    s2 += __shfl_xor(s2, off, 32);
  }
  const float mu  = s * (1.0f / 256.0f);
  float var = (s2 - 256.0f * mu * mu) * (1.0f / 255.0f);
  float sigma = sqrtf(fmaxf(var, 0.0f));
  sigma = fmaxf(sigma, 1e-6f);
  const float inv = 1.0f / sigma;
  #pragma unroll
  for (int u = 0; u < 8; ++u) {
    const int c = lane + u*32;
    p[c] = (v[u] - mu) * inv * lnw[c] + lnb[c];
  }
}

// ---------------- host-side launcher ---------------------------------------
extern "C" void kernel_launch(void* const* d_in, const int* in_sizes, int n_in,
                              void* d_out, int out_size, void* d_ws, size_t ws_size,
                              hipStream_t stream) {
  const float* x    = (const float*)d_in[0];
  const float* wihf = (const float*)d_in[1];
  const float* whhf = (const float*)d_in[2];
  const float* bihf = (const float*)d_in[3];
  const float* bhhf = (const float*)d_in[4];
  const float* wihb = (const float*)d_in[5];
  const float* whhb = (const float*)d_in[6];
  const float* bihb = (const float*)d_in[7];
  const float* bhhb = (const float*)d_in[8];
  const float* lnw  = (const float*)d_in[9];
  const float* lnb  = (const float*)d_in[10];

  char* ws = (char*)d_ws;
  size_t off = 0;
  bf16* xb   = (bf16*)(ws + off); off += (size_t)BT * DPAD * 2;   // 65.0 MB
  bf16* wfb  = (bf16*)(ws + off); off += (size_t)G3 * DPAD * 2;
  bf16* wbb  = (bf16*)(ws + off); off += (size_t)G3 * DPAD * 2;
  bf16* hfb  = (bf16*)(ws + off); off += (size_t)G3 * HID * 2;
  bf16* hbb  = (bf16*)(ws + off); off += (size_t)G3 * HID * 2;
  float* xpb = (float*)(ws + off); off += 2ull * BT * G3 * 4;     // 100.7 MB

  // 0) bf16 conversion (+ K zero-pad to multiple of 32)
  {
    size_t tot = (size_t)BT * DPAD;
    conv_bf16_pad<<<dim3((unsigned)((tot + 255) / 256)), 256, 0, stream>>>(
        x, xb, BT, DIN, DPAD);
    tot = (size_t)G3 * DPAD;
    conv_bf16_pad<<<dim3((unsigned)((tot + 255) / 256)), 256, 0, stream>>>(
        wihf, wfb, G3, DIN, DPAD);
    conv_bf16_pad<<<dim3((unsigned)((tot + 255) / 256)), 256, 0, stream>>>(
        wihb, wbb, G3, DIN, DPAD);
    tot = (size_t)G3 * HID;
    conv_bf16_pad<<<dim3((unsigned)((tot + 255) / 256)), 256, 0, stream>>>(
        whhf, hfb, G3, HID, HID);
    conv_bf16_pad<<<dim3((unsigned)((tot + 255) / 256)), 256, 0, stream>>>(
        whhb, hbb, G3, HID, HID);
  }

  // 1) input projections for both directions (bf16 WMMA, fp32 accumulate)
  xp_gemm<<<dim3(BT / 16, 3, 2), 256, 0, stream>>>(xb, wfb, wbb, bihf, bihb, xpb);

  // 2) persistent bidirectional GRU scan (w_hh resident in LDS)
  gru_scan<<<dim3(8), 256, SCAN_SMEM, stream>>>(hfb, hbb, bhhf, bhhb, xpb,
                                                (float*)d_out);

  // 3) LayerNorm in-place on d_out
  layernorm_rows<<<dim3(BT / 8), 256, 0, stream>>>((float*)d_out, lnw, lnb);
}
